// AudioTransformerMAE_Encoder_53678501266183
// MI455X (gfx1250) — matrix operand order repaired and checked by reading, not compiled, not run on hardware
//
#include <hip/hip_runtime.h>

// ---------------- model dims ----------------
constexpr int BB   = 32;        // batch
constexpr int FF   = 64;        // mel bins
constexpr int TT   = 2000;      // time
constexpr int DM   = 768;       // model dim
constexpr int LLAY = 4;         // layers
constexpr int HH   = 12;        // heads
constexpr int EE   = 14;        // experts
constexpr int KTOP = 4;         // top-k experts
constexpr int HID  = 2 * DM;    // 1536
constexpr int SHID = 4 * DM;    // 3072
constexpr int SS   = TT / 4;    // 500 tokens per sequence
constexpr int MTOK = BB * SS;   // 16000 tokens
constexpr int DH   = DM / HH;   // 64 head dim
constexpr int CK   = FF * 4;    // 256 conv patch length
constexpr float EPSLN = 1e-5f;

typedef __attribute__((ext_vector_type(16))) _Float16 v16h;
typedef __attribute__((ext_vector_type(8)))  _Float16 v8h;
typedef __attribute__((ext_vector_type(4)))  _Float16 v4h;
typedef __attribute__((ext_vector_type(8)))  float    v8f;

__device__ __forceinline__ float gelu_exact(float v) {
    return 0.5f * v * (1.0f + erff(v * 0.70710678118654752440f));
}

// ================= WMMA GEMM core =================
// C[M,N] = epilogue( A[M,K](f32) * Bt[N,K](f16) )  via f16 WMMA, f32 accum.
// Block tile 128x128, K-step 64 => 16 v_wmma_f32_16x16x32_f16 per barrier round.
// 256 threads = 8 waves (4x2 wave grid); each wave owns a 32x64 output tile.
// A converted f32->f16 during staging; B pre-packed f16 [N][K].
// Global loads for tile kb+64 are software-pipelined behind the WMMAs of kb.
// All K dims in this model (256/768/1536/3072) are multiples of 64.
#define BM 128
#define BN 128
#define BKT 64
#define LDSP 72   // padded LDS row stride in halfs (64 + 8)

__device__ __forceinline__ void gemm_block(
    const float* __restrict__ A, int lda,
    const _Float16* __restrict__ Bt, int ldbt,   // [N][K] f16, ldbt == K
    const float* __restrict__ bias,
    float* __restrict__ C, int ldc,
    const float* __restrict__ res,               // optional residual (C-shaped)
    const float* __restrict__ pos, int posS,     // optional pos-embed [posS, ldc]
    int M, int N, int K,
    int doGelu, int atomicMode, float scale)
{
    __shared__ _Float16 sA[BM][LDSP];   // [m][k]
    __shared__ _Float16 sB[BN][LDSP];   // [n][k]

    const int tid  = threadIdx.x;
    const int lane = tid & 31;
    const int wave = tid >> 5;
    const int waveRow = wave >> 1;      // 0..3
    const int waveCol = wave & 1;       // 0..1
    const int blockM = blockIdx.y * BM;
    const int blockN = blockIdx.x * BN;
    const int halfSel = lane >> 4;      // lane half 0/1
    const int l16 = lane & 15;

    v8f acc[2][4];
#pragma unroll
    for (int mi = 0; mi < 2; ++mi)
#pragma unroll
        for (int ni = 0; ni < 4; ++ni) {
            v8f z = {0.f,0.f,0.f,0.f,0.f,0.f,0.f,0.f};
            acc[mi][ni] = z;
        }

    float4 aReg[8];   // 128x64 f32 tile: 2048 float4, 8 per thread
    v8h    bReg[4];   // 128x64 f16 tile: 1024 v8h,    4 per thread

    auto loadTile = [&](int kb) {
#pragma unroll
        for (int i = 0; i < 8; ++i) {
            int e   = tid + i * 256;      // float4 chunk id
            int row = e >> 4;             // 0..127  (16 float4 per 64-wide row)
            int c4  = e & 15;             // k/4
            int gr  = blockM + row;
            int grc = (gr < M) ? gr : (M - 1);      // branchless clamp
            float4 v = *(const float4*)&A[(size_t)grc * lda + kb + c4 * 4];
            if (gr >= M) { v.x = 0.f; v.y = 0.f; v.z = 0.f; v.w = 0.f; }
            aReg[i] = v;
        }
#pragma unroll
        for (int i = 0; i < 4; ++i) {
            int e  = tid + i * 256;       // v8h chunk id
            int n  = e >> 3;              // 0..127  (8 v8h per 64-wide row)
            int c8 = e & 7;               // k/8
            bReg[i] = *(const v8h*)&Bt[(size_t)(blockN + n) * ldbt + kb + c8 * 8];
        }
    };
    auto storeTile = [&]() {
#pragma unroll
        for (int i = 0; i < 8; ++i) {
            int e   = tid + i * 256;
            int row = e >> 4;
            int c4  = e & 15;
            float4 v = aReg[i];
            v4h p = {(_Float16)v.x, (_Float16)v.y, (_Float16)v.z, (_Float16)v.w};
            *(v4h*)&sA[row][c4 * 4] = p;
        }
#pragma unroll
        for (int i = 0; i < 4; ++i) {
            int e  = tid + i * 256;
            int n  = e >> 3;
            int c8 = e & 7;
            *(v8h*)&sB[n][c8 * 8] = bReg[i];
        }
    };

    loadTile(0);
    for (int kb = 0; kb < K; kb += BKT) {
        storeTile();
        __syncthreads();
        if (kb + BKT < K) loadTile(kb + BKT);     // overlap next tile with math
        if (kb + 2 * BKT < K) {                   // L2 prefetch two tiles ahead
            int gr = blockM + (tid >> 1);
            if (gr < M) __builtin_prefetch(&A[(size_t)gr * lda + kb + 2 * BKT], 0, 1);
        }

        // ---- fragments + 16 WMMAs per wave (two k=32 sub-steps) ----
#pragma unroll
        for (int kk = 0; kk < BKT; kk += 32) {
#pragma unroll
            for (int mi = 0; mi < 2; ++mi) {
                // A 16x32 f16 layout: lanes 0-15 K=[0..7,16..23],
                //                     lanes 16-31 K=[8..15,24..31]
                int row = waveRow * 32 + mi * 16 + l16;
                v8h alo = *(const v8h*)&sA[row][kk + halfSel * 8];
                v8h ahi = *(const v8h*)&sA[row][kk + 16 + halfSel * 8];
                v16h af = __builtin_shufflevector(alo, ahi,
                            0,1,2,3,4,5,6,7,8,9,10,11,12,13,14,15);
#pragma unroll
                for (int ni = 0; ni < 4; ++ni) {
                    // B 32x16 f16 layout: lane col N=l16; lanes 0-15 K=[0..7,16..23], lanes 16-31 K=[8..15,24..31]
                    int col = waveCol * 64 + ni * 16 + l16;
                    v8h blo = *(const v8h*)&sB[col][kk + halfSel * 8];
                    v8h bhi = *(const v8h*)&sB[col][kk + 16 + halfSel * 8];
                    v16h bf = __builtin_shufflevector(blo, bhi,
                                0,1,2,3,4,5,6,7,8,9,10,11,12,13,14,15);
                    acc[mi][ni] = __builtin_amdgcn_wmma_f32_16x16x32_f16(
                        false, af, false, bf, (short)0, acc[mi][ni], false, false);
                }
            }
        }
        __syncthreads();
    }

    // ---- epilogue: C/D layout VGPR r -> M = halfSel*8 + r, N = l16 ----
#pragma unroll
    for (int mi = 0; mi < 2; ++mi)
#pragma unroll
        for (int ni = 0; ni < 4; ++ni)
#pragma unroll
            for (int r = 0; r < 8; ++r) {
                int rowg = blockM + waveRow * 32 + mi * 16 + halfSel * 8 + r;
                int colg = blockN + waveCol * 64 + ni * 16 + l16;
                if (rowg >= M) continue;
                float v = acc[mi][ni][r];
                if (bias) v += bias[colg];
                if (doGelu) v = gelu_exact(v);
                size_t co = (size_t)rowg * ldc + colg;
                if (atomicMode) {
                    atomicAdd(&C[co], scale * v);
                } else {
                    if (res) v += res[co];
                    if (pos) v += pos[(size_t)(rowg % posS) * ldc + colg];
                    C[co] = v;
                }
            }
}

// ---------------- GEMM wrappers ----------------
__global__ __launch_bounds__(256)
void gemm_kernel(const float* A, int lda, const _Float16* Bt, int ldbt,
                 const float* bias, float* C, int ldc,
                 const float* res, const float* pos, int posS,
                 int M, int N, int K, int doGelu)
{
    gemm_block(A, lda, Bt, ldbt, bias, C, ldc, res, pos, posS, M, N, K, doGelu, 0, 1.0f);
}

// hmid[b] = gelu(xn[b] @ exp_w1[idx[b,k]] + exp_b1[idx[b,k]])
__global__ __launch_bounds__(256)
void expert_gemm1_kernel(const float* xn, const _Float16* w1t, const float* b1,
                         float* hmid, const int* idx, int kSel)
{
    int b = blockIdx.z;
    int e = idx[b * KTOP + kSel];
    gemm_block(xn + (size_t)b * SS * DM, DM,
               w1t + (size_t)e * DM * HID, DM,
               b1 + (size_t)e * HID,
               hmid + (size_t)b * SS * HID, HID,
               nullptr, nullptr, 0,
               SS, HID, DM, /*gelu*/1, 0, 1.0f);
}

// y[b] += w[b,k] * (hmid[b] @ exp_w2[idx[b,k]] + exp_b2[idx[b,k]])
__global__ __launch_bounds__(256)
void expert_gemm2_kernel(const float* hmid, const _Float16* w2t, const float* b2,
                         float* y, const int* idx, const float* wts, int kSel)
{
    int b = blockIdx.z;
    int e = idx[b * KTOP + kSel];
    float sc = wts[b * KTOP + kSel];
    gemm_block(hmid + (size_t)b * SS * HID, HID,
               w2t + (size_t)e * HID * DM, HID,
               b2 + (size_t)e * DM,
               y + (size_t)b * SS * DM, DM,
               nullptr, nullptr, 0,
               SS, DM, HID, 0, /*atomic*/1, sc);
}

// ---------------- weight packing ----------------
// W[K][N] f32 -> Wt[N][K] f16 (tiled transpose through LDS).
// grid: (N/32, K/32, batch); block (32,8). blockIdx.z strides by K*N.
__global__ __launch_bounds__(256)
void packT_kernel(const float* __restrict__ W, _Float16* __restrict__ Wt, int K, int N)
{
    __shared__ float tile[32][33];
    const size_t zoff = (size_t)blockIdx.z * K * N;
    const float* Wz = W + zoff;
    _Float16* Wtz = Wt + zoff;
    int k0 = blockIdx.y * 32, n0 = blockIdx.x * 32;
    for (int i = threadIdx.y; i < 32; i += 8)
        tile[i][threadIdx.x] = Wz[(size_t)(k0 + i) * N + n0 + threadIdx.x];
    __syncthreads();
    for (int i = threadIdx.y; i < 32; i += 8)
        Wtz[(size_t)(n0 + i) * K + k0 + threadIdx.x] = (_Float16)tile[threadIdx.x][i];
}

__global__ void cvt_f16_kernel(const float* __restrict__ s, _Float16* __restrict__ d, int n)
{
    int i = blockIdx.x * 256 + threadIdx.x;
    if (i < n) d[i] = (_Float16)s[i];
}

// ---------------- patch-embed packing ----------------
__global__ void pack_x_kernel(const float* __restrict__ x, float* __restrict__ Apack)
{
    int m = blockIdx.x;           // token
    int c = threadIdx.x;          // f*4 + t
    int b = m / SS, s = m % SS;
    int f = c >> 2, t = c & 3;
    Apack[(size_t)m * CK + c] = x[((size_t)b * FF + f) * TT + s * 4 + t];
}

// ---------------- LayerNorm (vectorized float4) ----------------
__global__ __launch_bounds__(256)
void layernorm_kernel(const float* __restrict__ in, const float* __restrict__ sc,
                      const float* __restrict__ bi, float* __restrict__ out)
{
    __shared__ float r1[256], r2[256];
    const int row = blockIdx.x;
    const float4* x4 = (const float4*)(in + (size_t)row * DM);
    const int j = threadIdx.x;                       // valid chunks: 0..191
    float4 v = make_float4(0.f, 0.f, 0.f, 0.f);
    float s = 0.f, s2 = 0.f;
    if (j < DM / 4) {
        v = x4[j];
        s  = v.x + v.y + v.z + v.w;
        s2 = v.x * v.x + v.y * v.y + v.z * v.z + v.w * v.w;
    }
    r1[j] = s; r2[j] = s2;
    __syncthreads();
    for (int st = 128; st > 0; st >>= 1) {
        if (j < st) { r1[j] += r1[j + st]; r2[j] += r2[j + st]; }
        __syncthreads();
    }
    float mu  = r1[0] * (1.0f / DM);
    float var = r2[0] * (1.0f / DM) - mu * mu;
    float inv = rsqrtf(var + EPSLN);
    if (j < DM / 4) {
        float4 sv = ((const float4*)sc)[j];
        float4 bv = ((const float4*)bi)[j];
        float4 o;
        o.x = (v.x - mu) * inv * sv.x + bv.x;
        o.y = (v.y - mu) * inv * sv.y + bv.y;
        o.z = (v.z - mu) * inv * sv.z + bv.z;
        o.w = (v.w - mu) * inv * sv.w + bv.w;
        ((float4*)(out + (size_t)row * DM))[j] = o;
    }
}

// ---------------- attention (flash-style, K/V f16 in LDS) ----------------
__global__ __launch_bounds__(256)
void attn_kernel(const float* __restrict__ qkv, float* __restrict__ attno)
{
    extern __shared__ _Float16 smem[];
    _Float16* kh = smem;            // [SS*DH]
    _Float16* vh = smem + SS * DH;  // [SS*DH]
    const int h = blockIdx.x, b = blockIdx.y;
    const size_t baseTok = (size_t)b * SS;

    for (int e = threadIdx.x; e < SS * DH / 4; e += 256) {
        int s = e >> 4, d4 = e & 15;
        size_t o = (baseTok + s) * (size_t)(3 * DM) + h * DH + d4 * 4;
        float4 kf = *(const float4*)&qkv[o + DM];
        float4 vf = *(const float4*)&qkv[o + 2 * DM];
        v4h kp = {(_Float16)kf.x, (_Float16)kf.y, (_Float16)kf.z, (_Float16)kf.w};
        v4h vp = {(_Float16)vf.x, (_Float16)vf.y, (_Float16)vf.z, (_Float16)vf.w};
        *(v4h*)&kh[s * DH + d4 * 4] = kp;
        *(v4h*)&vh[s * DH + d4 * 4] = vp;
    }
    __syncthreads();

    const float scale = 0.125f;     // (D/H)^-0.5 = 1/8
    for (int q = threadIdx.x; q < SS; q += 256) {
        float qv[DH];
        size_t qo = (baseTok + q) * (size_t)(3 * DM) + h * DH;
#pragma unroll
        for (int c = 0; c < DH / 4; ++c) {
            float4 qf = *(const float4*)&qkv[qo + c * 4];
            qv[c * 4 + 0] = qf.x * scale;
            qv[c * 4 + 1] = qf.y * scale;
            qv[c * 4 + 2] = qf.z * scale;
            qv[c * 4 + 3] = qf.w * scale;
        }
        float m = -1e30f, l = 0.f;
        float acc[DH];
#pragma unroll
        for (int d = 0; d < DH; ++d) acc[d] = 0.f;
        for (int s = 0; s < SS; ++s) {
            const v8h* kr = (const v8h*)&kh[s * DH];
            float dot = 0.f;
#pragma unroll
            for (int c = 0; c < DH / 8; ++c) {
                v8h kk = kr[c];
#pragma unroll
                for (int jj = 0; jj < 8; ++jj) dot += qv[c * 8 + jj] * (float)kk[jj];
            }
            float mn = fmaxf(m, dot);
            float corr = __expf(m - mn);
            float p = __expf(dot - mn);
            l = l * corr + p;
            const v8h* vr = (const v8h*)&vh[s * DH];
#pragma unroll
            for (int c = 0; c < DH / 8; ++c) {
                v8h vv = vr[c];
#pragma unroll
                for (int jj = 0; jj < 8; ++jj)
                    acc[c * 8 + jj] = acc[c * 8 + jj] * corr + p * (float)vv[jj];
            }
            m = mn;
        }
        float inv = 1.0f / l;
        size_t oo = (baseTok + q) * (size_t)DM + h * DH;
#pragma unroll
        for (int c = 0; c < DH / 4; ++c) {
            float4 o;
            o.x = acc[c * 4 + 0] * inv;
            o.y = acc[c * 4 + 1] * inv;
            o.z = acc[c * 4 + 2] * inv;
            o.w = acc[c * 4 + 3] * inv;
            *(float4*)&attno[oo + c * 4] = o;
        }
    }
}

// ---------------- sequence mean over S (vectorized) ----------------
__global__ void seqmean_kernel(const float* __restrict__ xn, float* __restrict__ seq)
{
    int b = blockIdx.x, d4 = threadIdx.x;   // blockDim = 192 = DM/4
    float4 s = make_float4(0.f, 0.f, 0.f, 0.f);
    for (int t = 0; t < SS; ++t) {
        float4 v = *(const float4*)&xn[((size_t)b * SS + t) * DM + d4 * 4];
        s.x += v.x; s.y += v.y; s.z += v.z; s.w += v.w;
    }
    const float inv = 1.0f / SS;
    s.x *= inv; s.y *= inv; s.z *= inv; s.w *= inv;
    *(float4*)&seq[(size_t)b * DM + d4 * 4] = s;
}

// ---------------- MoE gating: gelu-MLP -> softmax -> top-4 -> softmax ----------------
__global__ __launch_bounds__(256)
void gate_kernel(const float* __restrict__ seq,
                 const float* __restrict__ w1, const float* __restrict__ b1,
                 const float* __restrict__ w2, const float* __restrict__ b2,
                 int* __restrict__ idx, float* __restrict__ wts)
{
    int b = blockIdx.x;
    __shared__ float sv[DM];
    __shared__ float g1[DM];
    __shared__ float logits[EE];
    for (int j = threadIdx.x; j < DM; j += 256) sv[j] = seq[(size_t)b * DM + j];
    __syncthreads();
    for (int j = threadIdx.x; j < DM; j += 256) {
        float acc = b1[j];
        for (int d = 0; d < DM; ++d) acc += sv[d] * w1[(size_t)d * DM + j];
        g1[j] = gelu_exact(acc);
    }
    __syncthreads();
    if (threadIdx.x < EE) {
        int j = threadIdx.x;
        float acc = b2[j];
        for (int d = 0; d < DM; ++d) acc += g1[d] * w2[(size_t)d * EE + j];
        logits[j] = acc;
    }
    __syncthreads();
    if (threadIdx.x == 0) {
        float mx = -1e30f;
        for (int e = 0; e < EE; ++e) mx = fmaxf(mx, logits[e]);
        float pr[EE]; float den = 0.f;
        for (int e = 0; e < EE; ++e) { pr[e] = __expf(logits[e] - mx); den += pr[e]; }
        float invd = 1.0f / den;
        for (int e = 0; e < EE; ++e) pr[e] *= invd;
        int sel[KTOP]; float pv[KTOP];
        for (int k = 0; k < KTOP; ++k) {
            int best = 0; float bv = -1.f;
            for (int e = 0; e < EE; ++e) {
                bool taken = false;
                for (int kk = 0; kk < k; ++kk) taken |= (sel[kk] == e);
                if (!taken && pr[e] > bv) { bv = pr[e]; best = e; }
            }
            sel[k] = best; pv[k] = bv;
        }
        float m2 = -1e30f;
        for (int k = 0; k < KTOP; ++k) m2 = fmaxf(m2, pv[k]);
        float d2 = 0.f; float wv[KTOP];
        for (int k = 0; k < KTOP; ++k) { wv[k] = __expf(pv[k] - m2); d2 += wv[k]; }
        float inv2 = 1.0f / d2;
        for (int k = 0; k < KTOP; ++k) {
            idx[b * KTOP + k] = sel[k];
            wts[b * KTOP + k] = wv[k] * inv2;
        }
    }
}

// ---------------- elementwise residual add (float4) ----------------
__global__ void add_inplace_kernel(float* __restrict__ dst, const float* __restrict__ src, int n4)
{
    int i = blockIdx.x * 256 + threadIdx.x;
    if (i < n4) {
        float4 a = ((const float4*)dst)[i];
        float4 b = ((const float4*)src)[i];
        a.x += b.x; a.y += b.y; a.z += b.z; a.w += b.w;
        ((float4*)dst)[i] = a;
    }
}

// =============================================================
extern "C" void kernel_launch(void* const* d_in, const int* in_sizes, int n_in,
                              void* d_out, int out_size, void* d_ws, size_t ws_size,
                              hipStream_t stream)
{
    const float* x        = (const float*)d_in[0];
    const float* conv_w   = (const float*)d_in[1];
    const float* conv_b   = (const float*)d_in[2];
    const float* pos      = (const float*)d_in[3];
    const float* ln1_s    = (const float*)d_in[4];
    const float* ln1_b    = (const float*)d_in[5];
    const float* w_qkv    = (const float*)d_in[6];
    const float* w_proj   = (const float*)d_in[7];
    const float* b_proj   = (const float*)d_in[8];
    const float* ln2_s    = (const float*)d_in[9];
    const float* ln2_b    = (const float*)d_in[10];
    const float* gate_w1  = (const float*)d_in[11];
    const float* gate_b1  = (const float*)d_in[12];
    const float* gate_w2  = (const float*)d_in[13];
    const float* gate_b2  = (const float*)d_in[14];
    const float* exp_w1   = (const float*)d_in[15];
    const float* exp_b1   = (const float*)d_in[16];
    const float* exp_w2   = (const float*)d_in[17];
    const float* exp_b2   = (const float*)d_in[18];
    const float* sh_w1    = (const float*)d_in[19];
    const float* sh_b1    = (const float*)d_in[20];
    const float* sh_w2    = (const float*)d_in[21];
    const float* sh_b2    = (const float*)d_in[22];
    const float* lnf_s    = (const float*)d_in[23];
    const float* lnf_b    = (const float*)d_in[24];

    // ---- workspace carve: f32 region ----
    float* w = (float*)d_ws;
    size_t off = 0;
    float* tok   = w + off; off += (size_t)MTOK * DM;
    float* xn    = w + off; off += (size_t)MTOK * DM;
    float* qkv   = w + off; off += (size_t)MTOK * 3 * DM;
    float* attno = w + off; off += (size_t)MTOK * DM;
    float* ybuf  = w + off; off += (size_t)MTOK * DM;
    float* smid  = w + off; off += (size_t)MTOK * SHID;
    float* hmid  = w + off; off += (size_t)BB * SS * HID;
    float* Apack = w + off; off += (size_t)MTOK * CK;
    float* seq   = w + off; off += (size_t)BB * DM;
    float* wts   = w + off; off += (size_t)BB * KTOP;
    int*   idx   = (int*)(w + off); off += (size_t)BB * KTOP;

    // ---- f16 region: transposed [N][K] weight packs ----
    _Float16* hb = (_Float16*)(w + off);
    size_t hoff = 0;
    _Float16* wt_conv = hb + hoff; hoff += (size_t)DM * CK;                 // [768][256]
    _Float16* wt_qkv  = hb + hoff; hoff += (size_t)LLAY * 3 * DM * DM;      // [2304][768] x4
    _Float16* wt_proj = hb + hoff; hoff += (size_t)LLAY * DM * DM;          // [768][768] x4
    _Float16* wt_sh1  = hb + hoff; hoff += (size_t)LLAY * SHID * DM;        // [3072][768] x4
    _Float16* wt_sh2  = hb + hoff; hoff += (size_t)LLAY * DM * SHID;        // [768][3072] x4
    _Float16* wt_e1   = hb + hoff; hoff += (size_t)LLAY * EE * HID * DM;    // [1536][768] x4x14
    _Float16* wt_e2   = hb + hoff; hoff += (size_t)LLAY * EE * DM * HID;    // [768][1536] x4x14
    (void)ws_size; (void)n_in; (void)in_sizes; (void)out_size; (void)hoff;

    const int MT125 = MTOK / BM;               // 125 (exact)
    const int ST4   = (SS + BM - 1) / BM;      // 4
    const size_t ATTN_LDS = (size_t)SS * DH * 2 * sizeof(_Float16); // 128000 B
    const dim3 tp(32, 8, 1);

    // ---- pack all weights to f16 [N][K] once per launch ----
    cvt_f16_kernel<<<(DM * CK + 255) / 256, 256, 0, stream>>>(conv_w, wt_conv, DM * CK);
    for (int i = 0; i < LLAY; ++i) {
        packT_kernel<<<dim3(3 * DM / 32, DM / 32, 1), tp, 0, stream>>>(
            w_qkv + (size_t)i * DM * 3 * DM, wt_qkv + (size_t)i * 3 * DM * DM, DM, 3 * DM);
        packT_kernel<<<dim3(DM / 32, DM / 32, 1), tp, 0, stream>>>(
            w_proj + (size_t)i * DM * DM, wt_proj + (size_t)i * DM * DM, DM, DM);
        packT_kernel<<<dim3(SHID / 32, DM / 32, 1), tp, 0, stream>>>(
            sh_w1 + (size_t)i * DM * SHID, wt_sh1 + (size_t)i * SHID * DM, DM, SHID);
        packT_kernel<<<dim3(DM / 32, SHID / 32, 1), tp, 0, stream>>>(
            sh_w2 + (size_t)i * SHID * DM, wt_sh2 + (size_t)i * DM * SHID, SHID, DM);
        packT_kernel<<<dim3(HID / 32, DM / 32, EE), tp, 0, stream>>>(
            exp_w1 + (size_t)i * EE * DM * HID, wt_e1 + (size_t)i * EE * HID * DM, DM, HID);
        packT_kernel<<<dim3(DM / 32, HID / 32, EE), tp, 0, stream>>>(
            exp_w2 + (size_t)i * EE * HID * DM, wt_e2 + (size_t)i * EE * DM * HID, HID, DM);
    }

    // ---- patch embed: pack + WMMA GEMM (+bias +pos) ----
    pack_x_kernel<<<MTOK, CK, 0, stream>>>(x, Apack);
    gemm_kernel<<<dim3(DM / BN, MT125, 1), 256, 0, stream>>>(
        Apack, CK, wt_conv, CK, conv_b, tok, DM,
        nullptr, pos, SS, MTOK, DM, CK, 0);

    for (int i = 0; i < LLAY; ++i) {
        const float* bp  = b_proj + (size_t)i * DM;
        const float* gw1 = gate_w1 + (size_t)i * DM * DM;
        const float* gb1 = gate_b1 + (size_t)i * DM;
        const float* gw2 = gate_w2 + (size_t)i * DM * EE;
        const float* gb2 = gate_b2 + (size_t)i * EE;
        const float* eb1 = exp_b1 + (size_t)i * EE * HID;
        const float* eb2 = exp_b2 + (size_t)i * EE * DM;
        const float* sb1 = sh_b1 + (size_t)i * SHID;
        const float* sb2 = sh_b2 + (size_t)i * DM;

        // --- attention block ---
        layernorm_kernel<<<MTOK, 256, 0, stream>>>(tok, ln1_s + (size_t)i * DM,
                                                   ln1_b + (size_t)i * DM, xn);
        gemm_kernel<<<dim3(3 * DM / BN, MT125, 1), 256, 0, stream>>>(
            xn, DM, wt_qkv + (size_t)i * 3 * DM * DM, DM, nullptr, qkv, 3 * DM,
            nullptr, nullptr, 0, MTOK, 3 * DM, DM, 0);
        attn_kernel<<<dim3(HH, BB, 1), 256, ATTN_LDS, stream>>>(qkv, attno);
        // tok = tok + attno @ w_proj + b_proj (residual fused in epilogue)
        gemm_kernel<<<dim3(DM / BN, MT125, 1), 256, 0, stream>>>(
            attno, DM, wt_proj + (size_t)i * DM * DM, DM, bp, tok, DM,
            tok, nullptr, 0, MTOK, DM, DM, 0);

        // --- MoE block (residual = tok) ---
        layernorm_kernel<<<MTOK, 256, 0, stream>>>(tok, ln2_s + (size_t)i * DM,
                                                   ln2_b + (size_t)i * DM, xn);
        seqmean_kernel<<<BB, DM / 4, 0, stream>>>(xn, seq);
        gate_kernel<<<BB, 256, 0, stream>>>(seq, gw1, gb1, gw2, gb2, idx, wts);

        // shared expert: y = gelu(xn @ sh_w1 + b) @ sh_w2 + b
        gemm_kernel<<<dim3(SHID / BN, MT125, 1), 256, 0, stream>>>(
            xn, DM, wt_sh1 + (size_t)i * SHID * DM, DM, sb1, smid, SHID,
            nullptr, nullptr, 0, MTOK, SHID, DM, 1);
        gemm_kernel<<<dim3(DM / BN, MT125, 1), 256, 0, stream>>>(
            smid, SHID, wt_sh2 + (size_t)i * DM * SHID, SHID, sb2, ybuf, DM,
            nullptr, nullptr, 0, MTOK, DM, SHID, 0);

        // routed experts: atomically accumulate w[b,k]*eout into y
        for (int k = 0; k < KTOP; ++k) {
            expert_gemm1_kernel<<<dim3(HID / BN, ST4, BB), 256, 0, stream>>>(
                xn, wt_e1 + (size_t)i * EE * HID * DM, eb1, hmid, idx, k);
            expert_gemm2_kernel<<<dim3(DM / BN, ST4, BB), 256, 0, stream>>>(
                hmid, wt_e2 + (size_t)i * EE * DM * HID, eb2, ybuf, idx, wts, k);
        }
        // tok = residual + y
        add_inplace_kernel<<<(MTOK * DM / 4 + 255) / 256, 256, 0, stream>>>(
            tok, ybuf, MTOK * DM / 4);
    }

    // final LayerNorm straight into d_out
    layernorm_kernel<<<MTOK, 256, 0, stream>>>(tok, lnf_s, lnf_b, (float*)d_out);
}
